// Attention_71279277244594
// MI455X (gfx1250) — compile-verified, matrix operand order checked by reading, not run on hardware
//
#include <hip/hip_runtime.h>

// ---------------------------------------------------------------------------
// MI455X / gfx1250 attention: bf16 WMMA (v_wmma_f32_16x16x32_bf16) everywhere.
// Prep kernels pre-convert x and weights to bf16 (weights transposed to
// N-major) so all GEMM staging is pure 16-byte copies.  If the toolchain
// exposes the gfx1250 async global->LDS builtins, GEMM tile staging uses
// GLOBAL_LOAD_ASYNC_TO_LDS_B128 (ASYNCcnt) instead of a VGPR round-trip.
// ---------------------------------------------------------------------------

typedef __attribute__((ext_vector_type(16))) __bf16 v16bf;
typedef __attribute__((ext_vector_type(8)))  __bf16 v8bf;
typedef __attribute__((ext_vector_type(8)))  float  v8f;
typedef __attribute__((ext_vector_type(4)))  float  v4f;
typedef __attribute__((ext_vector_type(4)))  int    v4i;

#if defined(__has_builtin)
#if __has_builtin(__builtin_amdgcn_global_load_async_to_lds_b128) && \
    __has_builtin(__builtin_amdgcn_s_wait_asynccnt)
#define HAVE_ASYNC_LDS 1
#endif
#endif

static constexpr int Bb    = 2;
static constexpr int Nn    = 2048;
static constexpr int QD    = 1024;
static constexpr int Hh    = 8;
static constexpr int HD    = 64;
static constexpr int INNER = 512;      // H*HD
static constexpr int BN    = Bb * Nn;  // 4096
static constexpr int LDP   = 80;       // padded LDS row (elems): 160B, 16B-aligned

__device__ __forceinline__ v8f wmma_bf16(v16bf a, v16bf b, v8f c) {
  // D = A(16x32) * B(32x16) + C(16x16), f32 accumulate
  return __builtin_amdgcn_wmma_f32_16x16x32_bf16(false, a, false, b,
                                                 (short)0, c, false, false);
}

// CDNA5 16-bit A/B fragment pattern for a K=32 chunk (ISA 7.12.2):
//   element e -> K = e + (e>=8 ? 8 : 0) + (lane>=16 ? 8 : 0)
// i.e. two contiguous 16-byte loads at byte offsets hi*16 and 32+hi*16.
__device__ __forceinline__ v16bf load_frag(const __bf16* p, int hi) {
  const v8bf lo = *(const v8bf*)(p + hi * 8);
  const v8bf hv = *(const v8bf*)(p + 16 + hi * 8);
  return __builtin_shufflevector(lo, hv, 0, 1, 2, 3, 4, 5, 6, 7,
                                         8, 9, 10, 11, 12, 13, 14, 15);
}

// 16B global->LDS tile copy: async path if available, else VGPR round-trip.
__device__ __forceinline__ void stage16(const __bf16* g, __bf16* l) {
#ifdef HAVE_ASYNC_LDS
  __builtin_amdgcn_global_load_async_to_lds_b128(
      (__attribute__((address_space(1))) v4i*)g,
      (__attribute__((address_space(3))) v4i*)l, 0, 0);
#else
  *(v8bf*)l = *(const v8bf*)g;
#endif
}
__device__ __forceinline__ void stage_fence() {
#ifdef HAVE_ASYNC_LDS
  __builtin_amdgcn_s_wait_asynccnt(0);
#endif
}

// Reductions across the 16-lane half that owns one row group of a C fragment.
__device__ __forceinline__ float redmax16(float v) {
  v = fmaxf(v, __shfl_xor(v, 1));
  v = fmaxf(v, __shfl_xor(v, 2));
  v = fmaxf(v, __shfl_xor(v, 4));
  v = fmaxf(v, __shfl_xor(v, 8));
  return v;
}
__device__ __forceinline__ float redsum16(float v) {
  v += __shfl_xor(v, 1);
  v += __shfl_xor(v, 2);
  v += __shfl_xor(v, 4);
  v += __shfl_xor(v, 8);
  return v;
}

// ---------------------------------------------------------------------------
// Prep 1: elementwise f32 -> bf16 (x).  8 elems / thread.
// ---------------------------------------------------------------------------
__global__ __launch_bounds__(256) void cvt_x_kernel(
    const float* __restrict__ in, __bf16* __restrict__ outb) {
  const size_t i = ((size_t)blockIdx.x * 256 + threadIdx.x) * 8;
  const v4f a = *(const v4f*)&in[i];
  const v4f b = *(const v4f*)&in[i + 4];
  v8bf o;
  o[0] = (__bf16)a.x; o[1] = (__bf16)a.y; o[2] = (__bf16)a.z; o[3] = (__bf16)a.w;
  o[4] = (__bf16)b.x; o[5] = (__bf16)b.y; o[6] = (__bf16)b.z; o[7] = (__bf16)b.w;
  *(v8bf*)&outb[i] = o;
}

// ---------------------------------------------------------------------------
// Prep 2: W[K][N] f32 -> Wt[N][K] bf16 (LDS-tiled 64x64 transpose).
// ---------------------------------------------------------------------------
__global__ __launch_bounds__(256) void transpose_w_kernel(
    const float* __restrict__ W, __bf16* __restrict__ Wt,
    int K, int N) {
  __shared__ __bf16 T[64][65];
  const int tid = threadIdx.x;
  const int k0 = blockIdx.x * 64;
  const int n0 = blockIdx.y * 64;
#pragma unroll
  for (int it = 0; it < 16; ++it) {
    const int lin = it * 256 + tid;
    const int r = lin >> 6, c = lin & 63;  // r: k-local, c: n-local
    T[r][c] = (__bf16)W[(size_t)(k0 + r) * N + n0 + c];
  }
  __syncthreads();
#pragma unroll
  for (int it = 0; it < 16; ++it) {
    const int lin = it * 256 + tid;
    const int n = lin >> 6, kk = lin & 63;
    Wt[(size_t)(n0 + n) * K + k0 + kk] = T[kk][n];
  }
}

// ---------------------------------------------------------------------------
// Kernel 1: fused QKV projection.  Y = xb @ W{q,k,v}, all-bf16 staging.
// 64x64 tile, K-step 64, 8 waves, 2 C-frags per wave.  blockIdx.z selects W.
// Q,K stored [B,H,N,HD]; V stored transposed [B,H,HD,N] for the PV stage.
// ---------------------------------------------------------------------------
__global__ __launch_bounds__(256) void qkv_proj_kernel(
    const __bf16* __restrict__ xb,                      // [BN, QD]
    const __bf16* __restrict__ WqT, const __bf16* __restrict__ WkT,
    const __bf16* __restrict__ WvT,                     // [INNER, QD]
    __bf16* __restrict__ Qb, __bf16* __restrict__ Kb,   // [B*H*N*HD]
    __bf16* __restrict__ Vt)                            // [B*H*HD*N]
{
  __shared__ __align__(16) __bf16 As[64][LDP];
  __shared__ __align__(16) __bf16 Bs[64][LDP];  // [col][k]

  const int which = blockIdx.z;
  const __bf16* WT = (which == 0) ? WqT : (which == 1) ? WkT : WvT;

  const int tid  = threadIdx.x;
  const int lane = tid & 31, wave = tid >> 5;
  const int l16  = lane & 15, hi = lane >> 4;
  const int rstrip = wave >> 1;   // 0..3 : 16-row strip
  const int chalf  = wave & 1;    // 0..1 : 32-col half
  const int m0 = blockIdx.y * 64;
  const int n0 = blockIdx.x * 64;

  v8f acc[2] = {v8f{}, v8f{}};

  for (int k0 = 0; k0 < QD; k0 += 64) {
    // pure 16B copies: 2 per thread for A, 2 for B
#pragma unroll
    for (int it = 0; it < 2; ++it) {
      const int lin = it * 2048 + tid * 8;
      const int r = lin >> 6, c = lin & 63;
      stage16(&xb[(size_t)(m0 + r) * QD + k0 + c], &As[r][c]);
    }
#pragma unroll
    for (int it = 0; it < 2; ++it) {
      const int lin = it * 2048 + tid * 8;
      const int cl = lin >> 6, kk = lin & 63;
      stage16(&WT[(size_t)(n0 + cl) * QD + k0 + kk], &Bs[cl][kk]);
    }
    stage_fence();
    __syncthreads();
#pragma unroll
    for (int kc = 0; kc < 2; ++kc) {
      const v16bf a  = load_frag(&As[rstrip * 16 + l16][kc * 32], hi);
      const v16bf b0 = load_frag(&Bs[chalf * 32 + l16][kc * 32], hi);
      const v16bf b1 = load_frag(&Bs[chalf * 32 + 16 + l16][kc * 32], hi);
      acc[0] = wmma_bf16(a, b0, acc[0]);
      acc[1] = wmma_bf16(a, b1, acc[1]);
    }
    __syncthreads();
  }

  // epilogue: C rows = e + hi*8, cols = l16 (+frag offsets)
  const int gm0 = m0 + rstrip * 16 + hi * 8;  // multiple of 8
  const int b   = gm0 >> 11;                  // rows gm0..gm0+7 share b
  const int nb  = gm0 & (Nn - 1);
#pragma unroll
  for (int f = 0; f < 2; ++f) {
    const int j = n0 + chalf * 32 + f * 16 + l16;  // 0..INNER-1
    const int h = j >> 6, d = j & 63;
    if (which == 2) {
      // Vt: consecutive e -> consecutive tokens -> one 16B store
      v8bf pk;
#pragma unroll
      for (int e = 0; e < 8; ++e) pk[e] = (__bf16)acc[f][e];
      *(v8bf*)&Vt[((size_t)(b * Hh + h) * HD + d) * Nn + nb] = pk;
    } else {
      __bf16* dst = ((which == 0) ? Qb : Kb) +
                    ((size_t)(b * Hh + h) * Nn + nb) * HD + d;
#pragma unroll
      for (int e = 0; e < 8; ++e) dst[(size_t)e * HD] = (__bf16)acc[f][e];
    }
  }
}

// ---------------------------------------------------------------------------
// Kernel 2: streaming-softmax attention.  One wave = one 16-query block of
// one (b,h).  Q/K fragments loaded straight from global in WMMA layout;
// P re-laid-out C->A through a private LDS tile; PV accumulated over 32-key
// chunks with online-softmax rescale.
// ---------------------------------------------------------------------------
__global__ __launch_bounds__(256) void attn_kernel(
    const __bf16* __restrict__ Qb, const __bf16* __restrict__ Kb,
    const __bf16* __restrict__ Vt, __bf16* __restrict__ Ob /* [BN, INNER] */)
{
  __shared__ __align__(16) __bf16 Plds[8][16][40];  // padded rows (80B)

  const int tid  = threadIdx.x;
  const int lane = tid & 31, wave = tid >> 5;
  const int l16  = lane & 15, hi = lane >> 4;
  const int task = blockIdx.x * 8 + wave;  // 0..2047
  const int bh = task >> 7;                // 0..15  (b*H + h)
  const int qt = task & 127;
  const int qbase = qt * 16;

  const __bf16* Qrow = Qb + ((size_t)bh * Nn + qbase + l16) * HD;
  const v16bf qa0 = load_frag(Qrow, hi);       // d = 0..31
  const v16bf qa1 = load_frag(Qrow + 32, hi);  // d = 32..63

  v8f O[4] = {v8f{}, v8f{}, v8f{}, v8f{}};
  float mrun[8], lrun[8];
#pragma unroll
  for (int e = 0; e < 8; ++e) { mrun[e] = -1e30f; lrun[e] = 0.f; }

  const float kscale = 0.125f * 1.44269504088896f;  // HD^-0.5 * log2(e)

  for (int kb = 0; kb < Nn; kb += 32) {
    // ---- scores: S(16x32) = Q(16x64) * K^T(64x32), two 16x16 C-frags ----
    v8f S0 = {}, S1 = {};
    const __bf16* Krow0 = Kb + ((size_t)bh * Nn + kb + l16) * HD;
    const __bf16* Krow1 = Krow0 + 16 * HD;
    if (kb + 32 < Nn) {  // uniform branch: prefetch next key block
      __builtin_prefetch(Krow0 + 32 * HD, 0, 0);
    }
    {
      const v16bf b0 = load_frag(Krow0, hi);
      const v16bf b1 = load_frag(Krow0 + 32, hi);
      const v16bf c0 = load_frag(Krow1, hi);
      const v16bf c1 = load_frag(Krow1 + 32, hi);
      S0 = wmma_bf16(qa0, b0, S0);
      S0 = wmma_bf16(qa1, b1, S0);
      S1 = wmma_bf16(qa0, c0, S1);
      S1 = wmma_bf16(qa1, c1, S1);
    }
    // ---- online softmax in log2 domain; C-frag row = e + hi*8 ----
    float corr[8];
#pragma unroll
    for (int e = 0; e < 8; ++e) {
      const float t0 = S0[e] * kscale;
      const float t1 = S1[e] * kscale;
      const float mt   = redmax16(fmaxf(t0, t1));
      const float mnew = fmaxf(mrun[e], mt);
      const float p0 = exp2f(t0 - mnew);
      const float p1 = exp2f(t1 - mnew);
      const float c  = exp2f(mrun[e] - mnew);
      const float rs = redsum16(p0 + p1);
      lrun[e] = lrun[e] * c + rs;
      mrun[e] = mnew;
      corr[e] = c;
      Plds[wave][e + hi * 8][l16]      = (__bf16)p0;
      Plds[wave][e + hi * 8][16 + l16] = (__bf16)p1;
    }
    // DS ops from one wave are processed in order; load sees the stores.
    const v16bf pa = load_frag(&Plds[wave][l16][0], hi);
    // ---- O(16x64) += P(16x32) * V(32x64), with per-row rescale ----
#pragma unroll
    for (int dt = 0; dt < 4; ++dt) {
      const __bf16* Vrow =
          Vt + ((size_t)bh * HD + dt * 16 + l16) * Nn + kb;
      const v16bf vb = load_frag(Vrow, hi);
#pragma unroll
      for (int e = 0; e < 8; ++e) O[dt][e] *= corr[e];
      O[dt] = wmma_bf16(pa, vb, O[dt]);
    }
  }

  // ---- finalize: O /= l ; write bf16 attn output [BN, INNER] ----
  const int b = bh >> 3, h = bh & 7;
#pragma unroll
  for (int e = 0; e < 8; ++e) {
    const float inv = 1.0f / lrun[e];
    const int n = qbase + e + hi * 8;
    __bf16* dst = Ob + ((size_t)b * Nn + n) * INNER + h * HD + l16;
    dst[0]  = (__bf16)(O[0][e] * inv);
    dst[16] = (__bf16)(O[1][e] * inv);
    dst[32] = (__bf16)(O[2][e] * inv);
    dst[48] = (__bf16)(O[3][e] * inv);
  }
}

// ---------------------------------------------------------------------------
// Kernel 3: output projection.  out = attn(bf16 [BN,INNER]) @ Wout + bout, f32.
// ---------------------------------------------------------------------------
__global__ __launch_bounds__(256) void out_proj_kernel(
    const __bf16* __restrict__ Ain,    // [BN, INNER]
    const __bf16* __restrict__ WoT,    // [QD, INNER]  (transposed, bf16)
    const float* __restrict__ bout,    // [QD]
    float* __restrict__ out)           // [BN, QD]
{
  __shared__ __align__(16) __bf16 As[64][LDP];
  __shared__ __align__(16) __bf16 Bs[64][LDP];

  const int tid  = threadIdx.x;
  const int lane = tid & 31, wave = tid >> 5;
  const int l16  = lane & 15, hi = lane >> 4;
  const int rstrip = wave >> 1;
  const int chalf  = wave & 1;
  const int m0 = blockIdx.y * 64;
  const int n0 = blockIdx.x * 64;

  v8f acc[2] = {v8f{}, v8f{}};

  for (int k0 = 0; k0 < INNER; k0 += 64) {
#pragma unroll
    for (int it = 0; it < 2; ++it) {
      const int lin = it * 2048 + tid * 8;
      const int r = lin >> 6, c = lin & 63;
      stage16(&Ain[(size_t)(m0 + r) * INNER + k0 + c], &As[r][c]);
    }
#pragma unroll
    for (int it = 0; it < 2; ++it) {
      const int lin = it * 2048 + tid * 8;
      const int cl = lin >> 6, kk = lin & 63;
      stage16(&WoT[(size_t)(n0 + cl) * INNER + k0 + kk], &Bs[cl][kk]);
    }
    stage_fence();
    __syncthreads();
#pragma unroll
    for (int kc = 0; kc < 2; ++kc) {
      const v16bf a  = load_frag(&As[rstrip * 16 + l16][kc * 32], hi);
      const v16bf b0 = load_frag(&Bs[chalf * 32 + l16][kc * 32], hi);
      const v16bf b1 = load_frag(&Bs[chalf * 32 + 16 + l16][kc * 32], hi);
      acc[0] = wmma_bf16(a, b0, acc[0]);
      acc[1] = wmma_bf16(a, b1, acc[1]);
    }
    __syncthreads();
  }

  const int gm0 = m0 + rstrip * 16 + hi * 8;
#pragma unroll
  for (int f = 0; f < 2; ++f) {
    const int j  = n0 + chalf * 32 + f * 16 + l16;
    const float bj = bout[j];
#pragma unroll
    for (int e = 0; e < 8; ++e)
      out[(size_t)(gm0 + e) * QD + j] = acc[f][e] + bj;
  }
}

// ---------------------------------------------------------------------------
extern "C" void kernel_launch(void* const* d_in, const int* in_sizes, int n_in,
                              void* d_out, int out_size, void* d_ws,
                              size_t ws_size, hipStream_t stream) {
  const float* x    = (const float*)d_in[0];
  const float* Wq   = (const float*)d_in[1];
  const float* Wk   = (const float*)d_in[2];
  const float* Wv   = (const float*)d_in[3];
  const float* Wout = (const float*)d_in[4];
  const float* bout = (const float*)d_in[5];
  float* out = (float*)d_out;

  char* ws = (char*)d_ws;
  const size_t MB = 1024 * 1024;
  __bf16* Qb  = (__bf16*)(ws + 0 * MB);   // 4 MiB  [B,H,N,HD]
  __bf16* Kb  = (__bf16*)(ws + 4 * MB);   // 4 MiB  [B,H,N,HD]
  __bf16* Vt  = (__bf16*)(ws + 8 * MB);   // 4 MiB  [B,H,HD,N]
  __bf16* Ob  = (__bf16*)(ws + 12 * MB);  // 4 MiB  [BN, INNER]
  __bf16* xb  = (__bf16*)(ws + 16 * MB);  // 8 MiB  [BN, QD]
  __bf16* WqT = (__bf16*)(ws + 24 * MB);  // 1 MiB  [INNER, QD]
  __bf16* WkT = (__bf16*)(ws + 25 * MB);  // 1 MiB
  __bf16* WvT = (__bf16*)(ws + 26 * MB);  // 1 MiB
  __bf16* WoT = (__bf16*)(ws + 27 * MB);  // 1 MiB  [QD, INNER]

  dim3 blk(256);
  // prep: conversions + transposes
  cvt_x_kernel<<<dim3((BN * QD) / (256 * 8)), blk, 0, stream>>>(x, xb);
  transpose_w_kernel<<<dim3(QD / 64, INNER / 64), blk, 0, stream>>>(
      Wq, WqT, QD, INNER);
  transpose_w_kernel<<<dim3(QD / 64, INNER / 64), blk, 0, stream>>>(
      Wk, WkT, QD, INNER);
  transpose_w_kernel<<<dim3(QD / 64, INNER / 64), blk, 0, stream>>>(
      Wv, WvT, QD, INNER);
  transpose_w_kernel<<<dim3(INNER / 64, QD / 64), blk, 0, stream>>>(
      Wout, WoT, INNER, QD);
  // main pipeline
  qkv_proj_kernel<<<dim3(INNER / 64, BN / 64, 3), blk, 0, stream>>>(
      xb, WqT, WkT, WvT, Qb, Kb, Vt);
  attn_kernel<<<dim3((Bb * Hh * (Nn / 16)) / 8), blk, 0, stream>>>(
      Qb, Kb, Vt, Ob);
  out_proj_kernel<<<dim3(QD / 64, BN / 64), blk, 0, stream>>>(
      Ob, WoT, bout, out);
}
